// GritRoPEPairTransformerLayer_23880018166294
// MI455X (gfx1250) — matrix operand order repaired and checked by reading, not compiled, not run on hardware
//
#include <hip/hip_runtime.h>
#include <math.h>

#define DIM   256
#define HIDE  512     // edge-MLP hidden (= FFN hidden)
#define IN3   768
#define NNODE 50000
#define NEDGE 250000
#define EPSV  1e-5f

typedef __attribute__((ext_vector_type(2)))  float    v2f;
typedef __attribute__((ext_vector_type(8)))  float    v8f;
typedef __attribute__((ext_vector_type(8)))  _Float16 v8h;
typedef __attribute__((ext_vector_type(16))) _Float16 v16h;

__device__ __forceinline__ v16h cat8(v8h lo, v8h hi) {
  return __builtin_shufflevector(lo, hi, 0, 1, 2, 3, 4, 5, 6, 7,
                                         8, 9, 10, 11, 12, 13, 14, 15);
}

// ---- wave-level f16 WMMA GEMM: T output tiles (16x16 each) per wave ------------
// A (f16, LDS): 16 x K row-major, lda = LDA halves.
// W (f16, global): row-major [N x K], ldw halves; B[k][n] = W[n][k].
// Tile t covers columns n in [n_base0 + t*128, +16).
// Fragment layouts per CDNA5 ISA 7.12.2 (16-bit, wave32):
//   A: lanes 0-15 m=lane  K {k0+0..7} in elems0-7, {k0+16..23} in elems8-15
//      lanes16-31 m=lane-16, K ranges shifted by +8.
//   B: same K split as A (lanes 0-15 n=lane, lanes16-31 n=lane-16).
//   C/D: vgpr i: lanes0-15 (M=i,N=lane); lanes16-31 (M=i+8,N=lane-16).
template <int K, int LDA, int T>
__device__ __forceinline__ void wave_gemm_multi(const _Float16* __restrict__ A,
                                                const _Float16* __restrict__ W,
                                                int ldw, int n_base0, v8f* acc) {
  const int lane = threadIdx.x & 31;
  const int l16  = lane & 15;
  const int half = lane >> 4;
  const _Float16* __restrict__ arow = A + l16 * LDA;
  #pragma unroll 2
  for (int k = 0; k < K; k += 32) {
    v8h alo = *(const v8h*)(arow + k + half * 8);
    v8h ahi = *(const v8h*)(arow + k + 16 + half * 8);
    v16h a = cat8(alo, ahi);
    #pragma unroll
    for (int t = 0; t < T; ++t) {
      const _Float16* __restrict__ wrow =
          W + (size_t)(n_base0 + t * 128 + l16) * ldw + k + half * 8;
      v16h b = cat8(*(const v8h*)wrow, *(const v8h*)(wrow + 16));
      acc[t] = __builtin_amdgcn_wmma_f32_16x16x32_f16(false, a, false, b,
                                                      (short)0, acc[t], false, false);
    }
  }
}

// ---------------- weight f32 -> f16 conversion -----------------------------------
__global__ void __launch_bounds__(256) k_cvt_f16(const float* __restrict__ src,
                                                 _Float16* __restrict__ dst, int n) {
  int i = blockIdx.x * 256 + threadIdx.x;
  if (i < n) dst[i] = (_Float16)src[i];
}

// ---------------- Kernel 1: node degree-scale + O-projection + residual ----------
__global__ void __launch_bounds__(256) k_node_proj(
    const float* __restrict__ x, const float* __restrict__ ha,
    const float* __restrict__ logdeg, const float* __restrict__ dc,
    const _Float16* __restrict__ WOh, const float* __restrict__ bO,
    float* __restrict__ hres) {
  __shared__ __align__(32) _Float16 At[16 * DIM];
  const int m0  = blockIdx.x * 16;
  const int tid = threadIdx.x;
  for (int f = tid; f < 16 * DIM; f += 256) {
    int r = f >> 8, c = f & 255;
    int m = m0 + r;
    float v  = ha[(size_t)m * DIM + c];
    float ld = logdeg[m];
    At[f] = (_Float16)(v * (dc[2 * c] + ld * dc[2 * c + 1]));
  }
  __syncthreads();
  const int wave = tid >> 5, lane = tid & 31;
  const int l16 = lane & 15, half = lane >> 4;
  v8f acc[2] = {};
  wave_gemm_multi<DIM, DIM, 2>(At, WOh, DIM, wave * 16, acc);
  #pragma unroll
  for (int t = 0; t < 2; ++t) {
    const int n = wave * 16 + t * 128 + l16;
    const float bias = bO[n];
    #pragma unroll
    for (int i = 0; i < 8; ++i) {
      int m = m0 + i + half * 8;
      size_t idx = (size_t)m * DIM + n;
      hres[idx] = x[idx] + acc[t][i] + bias;
    }
  }
}

// ---------------- Column statistics (sum / sumsq over rows, 256 features) --------
__global__ void __launch_bounds__(256) k_colstats(
    const float* __restrict__ src, int rows, int rows_per_block,
    float* __restrict__ sum, float* __restrict__ sumsq) {
  const int f  = threadIdx.x;
  int r0 = blockIdx.x * rows_per_block;
  int r1 = r0 + rows_per_block; if (r1 > rows) r1 = rows;
  float s = 0.f, q = 0.f;
  for (int r = r0; r < r1; ++r) {
    float v = src[(size_t)r * DIM + f];
    s += v; q += v * v;
  }
  atomicAdd(&sum[f], s);
  atomicAdd(&sumsq[f], q);
}

__global__ void k_bnfinal(const float* __restrict__ sum, const float* __restrict__ sumsq,
                          float inv_rows, float* __restrict__ mean, float* __restrict__ rstd) {
  const int f = threadIdx.x;     // 256 threads
  float m = sum[f] * inv_rows;
  float v = sumsq[f] * inv_rows - m * m;
  mean[f] = m;
  rstd[f] = rsqrtf(v + EPSV);
}

// ---------------- BN1h-normalize + FFN + residual --------------------------------
__global__ void __launch_bounds__(256) k_ffn(
    const float* __restrict__ hres,
    const float* __restrict__ mean, const float* __restrict__ rstd,
    const float* __restrict__ g, const float* __restrict__ b,
    const _Float16* __restrict__ Wf1h, const float* __restrict__ bf1,
    const _Float16* __restrict__ Wf2h, const float* __restrict__ bf2,
    float* __restrict__ hsum) {
  __shared__ float Htf[16 * DIM];
  __shared__ __align__(32) _Float16 Hth[16 * DIM];
  __shared__ __align__(32) _Float16 MidH[16 * HIDE];
  const int m0  = blockIdx.x * 16;
  const int tid = threadIdx.x;
  for (int f = tid; f < 16 * DIM; f += 256) {
    int r = f >> 8, c = f & 255;
    float v = hres[(size_t)(m0 + r) * DIM + c];
    float hn = (v - mean[c]) * rstd[c] * g[c] + b[c];
    Htf[f] = hn;
    Hth[f] = (_Float16)hn;
  }
  __syncthreads();
  const int wave = tid >> 5, lane = tid & 31;
  const int l16 = lane & 15, half = lane >> 4;
  {
    v8f acc[4] = {};
    wave_gemm_multi<DIM, DIM, 4>(Hth, Wf1h, DIM, wave * 16, acc);   // 256 -> 512
    #pragma unroll
    for (int t = 0; t < 4; ++t) {
      const int n = wave * 16 + t * 128 + l16;
      const float bias = bf1[n];
      #pragma unroll
      for (int i = 0; i < 8; ++i) {
        float v = acc[t][i] + bias;
        MidH[(i + half * 8) * HIDE + n] = (_Float16)(v > 0.f ? v : 0.f);
      }
    }
  }
  __syncthreads();
  {
    v8f acc[2] = {};
    wave_gemm_multi<HIDE, HIDE, 2>(MidH, Wf2h, HIDE, wave * 16, acc); // 512 -> 256
    #pragma unroll
    for (int t = 0; t < 2; ++t) {
      const int n = wave * 16 + t * 128 + l16;
      const float bias = bf2[n];
      #pragma unroll
      for (int i = 0; i < 8; ++i) {
        int r = i + half * 8;
        hsum[(size_t)(m0 + r) * DIM + n] = Htf[r * DIM + n] + acc[t][i] + bias;
      }
    }
  }
}

// ---------------- BN apply, float4-vectorized ------------------------------------
__global__ void __launch_bounds__(256) k_bnapply4(
    const float4* __restrict__ in, float4* __restrict__ out, size_t total4,
    const float* __restrict__ mean, const float* __restrict__ rstd,
    const float* __restrict__ g, const float* __restrict__ b) {
  size_t i = (size_t)blockIdx.x * blockDim.x + threadIdx.x;
  size_t stride = (size_t)gridDim.x * blockDim.x;
  for (; i < total4; i += stride) {
    int f = (int)((i & 63) << 2);          // feature of component .x (DIM=256)
    float4 v = in[i];
    float4 o;
    o.x = (v.x - mean[f + 0]) * rstd[f + 0] * g[f + 0] + b[f + 0];
    o.y = (v.y - mean[f + 1]) * rstd[f + 1] * g[f + 1] + b[f + 1];
    o.z = (v.z - mean[f + 2]) * rstd[f + 2] * g[f + 2] + b[f + 2];
    o.w = (v.w - mean[f + 3]) * rstd[f + 3] * g[f + 3] + b[f + 3];
    out[i] = o;
  }
}

// ---------------- Fused edge kernel ----------------------------------------------
// pair = LN([x[s]+x[d] | x[s]*x[d] | ea]); e_pre = ea + gelu(pair@W1^T+b1)@W2^T + b2
__global__ void __launch_bounds__(256) k_edge(
    const float* __restrict__ x, const float* __restrict__ ea,
    const int* __restrict__ ei,
    const float* __restrict__ lng, const float* __restrict__ lnb,
    const _Float16* __restrict__ W1h, const float* __restrict__ b1,
    const _Float16* __restrict__ W2h, const float* __restrict__ b2,
    float* __restrict__ eout) {
  __shared__ float PairF[16 * IN3];                 // 48 KB (f32 for LN stats)
  __shared__ __align__(32) _Float16 PairH[16 * IN3]; // 24 KB (GEMM A operand)
  __shared__ __align__(32) _Float16 MidH[16 * HIDE]; // 16 KB
  __shared__ int sidx[16], didx[16];
  const int e0  = blockIdx.x * 16;
  const int tid = threadIdx.x;
  if (tid < 16)       sidx[tid]      = ei[e0 + tid];
  else if (tid < 32)  didx[tid - 16] = ei[NEDGE + e0 + (tid - 16)];
  __syncthreads();
  for (int f = tid; f < 16 * DIM; f += 256) {
    int r = f >> 8, c = f & 255;
    float a  = x[(size_t)sidx[r] * DIM + c];
    float bv = x[(size_t)didx[r] * DIM + c];
    PairF[r * IN3 + c]           = a + bv;
    PairF[r * IN3 + DIM + c]     = a * bv;
    PairF[r * IN3 + 2 * DIM + c] = ea[(size_t)(e0 + r) * DIM + c];
  }
  __syncthreads();
  const int wave = tid >> 5, lane = tid & 31;
  // LayerNorm over 768, one wave per 2 rows; normalized result stored as f16
  #pragma unroll
  for (int rr = 0; rr < 2; ++rr) {
    const int r = wave * 2 + rr;
    float s = 0.f, q = 0.f;
    for (int c = lane; c < IN3; c += 32) {
      float v = PairF[r * IN3 + c];
      s += v; q += v * v;
    }
    #pragma unroll
    for (int off = 16; off > 0; off >>= 1) {
      s += __shfl_down(s, off, 32);
      q += __shfl_down(q, off, 32);
    }
    s = __shfl(s, 0, 32);
    q = __shfl(q, 0, 32);
    const float m  = s * (1.0f / IN3);
    const float vv = q * (1.0f / IN3) - m * m;
    const float rs = rsqrtf(vv + EPSV);
    for (int c = lane; c < IN3; c += 32) {
      float v = PairF[r * IN3 + c];
      PairH[r * IN3 + c] = (_Float16)((v - m) * rs * lng[c] + lnb[c]);
    }
  }
  __syncthreads();
  const int l16 = lane & 15, half = lane >> 4;
  {
    v8f acc[4] = {};
    wave_gemm_multi<IN3, IN3, 4>(PairH, W1h, IN3, wave * 16, acc);   // 768 -> 512
    #pragma unroll
    for (int t = 0; t < 4; ++t) {
      const int n = wave * 16 + t * 128 + l16;
      const float bias = b1[n];
      #pragma unroll
      for (int i = 0; i < 8; ++i) {
        float v = acc[t][i] + bias;   // exact GELU
        MidH[(i + half * 8) * HIDE + n] =
            (_Float16)(0.5f * v * (1.0f + erff(v * 0.70710678118654752440f)));
      }
    }
  }
  __syncthreads();
  {
    v8f acc[2] = {};
    wave_gemm_multi<HIDE, HIDE, 2>(MidH, W2h, HIDE, wave * 16, acc); // 512 -> 256
    #pragma unroll
    for (int t = 0; t < 2; ++t) {
      const int n = wave * 16 + t * 128 + l16;
      const float bias = b2[n];
      #pragma unroll
      for (int i = 0; i < 8; ++i) {
        int r = i + half * 8;
        size_t idx = (size_t)(e0 + r) * DIM + n;
        eout[idx] = ea[idx] + acc[t][i] + bias;
      }
    }
  }
}

// ---------------------------------------------------------------------------------
extern "C" void kernel_launch(void* const* d_in, const int* in_sizes, int n_in,
                              void* d_out, int out_size, void* d_ws, size_t ws_size,
                              hipStream_t stream) {
  (void)in_sizes; (void)n_in; (void)out_size; (void)ws_size;
  const float* x    = (const float*)d_in[0];
  const float* ea   = (const float*)d_in[1];
  const float* ha   = (const float*)d_in[2];
  const float* ldg  = (const float*)d_in[3];
  const int*   ei   = (const int*)  d_in[4];
  const float* dc   = (const float*)d_in[5];
  const float* WO   = (const float*)d_in[6];
  const float* bO   = (const float*)d_in[7];
  const float* lng  = (const float*)d_in[8];
  const float* lnb  = (const float*)d_in[9];
  const float* W1e  = (const float*)d_in[10];
  const float* b1e  = (const float*)d_in[11];
  const float* W2e  = (const float*)d_in[12];
  const float* b2e  = (const float*)d_in[13];
  const float* g1h  = (const float*)d_in[14];
  const float* bb1h = (const float*)d_in[15];
  const float* g1e  = (const float*)d_in[16];
  const float* bb1e = (const float*)d_in[17];
  const float* Wf1  = (const float*)d_in[18];
  const float* bf1  = (const float*)d_in[19];
  const float* Wf2  = (const float*)d_in[20];
  const float* bf2  = (const float*)d_in[21];
  const float* g2h  = (const float*)d_in[22];
  const float* bb2h = (const float*)d_in[23];

  float* out_h = (float*)d_out;                        // N x 256
  float* out_e = out_h + (size_t)NNODE * DIM;          // E x 256
  float* hsum  = out_e;   // scratch: reuse edge output region before edge phase

  // Workspace: [0, 16KB) BN stats (f32), [16KB, ...) f16 weight copies
  float* st = (float*)d_ws;
  float *s1h_sum = st,        *s1h_sq = st + 256,  *s1h_m = st + 512,  *s1h_r = st + 768;
  float *s2h_sum = st + 1024, *s2h_sq = st + 1280, *s2h_m = st + 1536, *s2h_r = st + 1792;
  float *s1e_sum = st + 2048, *s1e_sq = st + 2304, *s1e_m = st + 2560, *s1e_r = st + 2816;
  hipMemsetAsync(st, 0, 3072 * sizeof(float), stream);

  _Float16* WOh  = (_Float16*)((char*)d_ws + 16384);
  _Float16* W1h  = WOh  + (size_t)DIM  * DIM;    // 65536
  _Float16* W2h  = W1h  + (size_t)HIDE * IN3;    // 393216
  _Float16* Wf1h = W2h  + (size_t)DIM  * HIDE;   // 131072
  _Float16* Wf2h = Wf1h + (size_t)HIDE * DIM;    // 131072
  const int nWO = DIM * DIM, nW1 = HIDE * IN3, nW2 = DIM * HIDE,
            nF1 = HIDE * DIM, nF2 = DIM * HIDE;
  k_cvt_f16<<<(nWO + 255) / 256, 256, 0, stream>>>(WO,  WOh,  nWO);
  k_cvt_f16<<<(nW1 + 255) / 256, 256, 0, stream>>>(W1e, W1h,  nW1);
  k_cvt_f16<<<(nW2 + 255) / 256, 256, 0, stream>>>(W2e, W2h,  nW2);
  k_cvt_f16<<<(nF1 + 255) / 256, 256, 0, stream>>>(Wf1, Wf1h, nF1);
  k_cvt_f16<<<(nF2 + 255) / 256, 256, 0, stream>>>(Wf2, Wf2h, nF2);

  const int RPB = 256;
  // ---- node path ----
  k_node_proj<<<NNODE / 16, 256, 0, stream>>>(x, ha, ldg, dc, WOh, bO, out_h);
  k_colstats<<<(NNODE + RPB - 1) / RPB, 256, 0, stream>>>(out_h, NNODE, RPB, s1h_sum, s1h_sq);
  k_bnfinal<<<1, 256, 0, stream>>>(s1h_sum, s1h_sq, 1.0f / NNODE, s1h_m, s1h_r);
  k_ffn<<<NNODE / 16, 256, 0, stream>>>(out_h, s1h_m, s1h_r, g1h, bb1h,
                                        Wf1h, bf1, Wf2h, bf2, hsum);
  k_colstats<<<(NNODE + RPB - 1) / RPB, 256, 0, stream>>>(hsum, NNODE, RPB, s2h_sum, s2h_sq);
  k_bnfinal<<<1, 256, 0, stream>>>(s2h_sum, s2h_sq, 1.0f / NNODE, s2h_m, s2h_r);
  k_bnapply4<<<2048, 256, 0, stream>>>((const float4*)hsum, (float4*)out_h,
                                       (size_t)NNODE * DIM / 4, s2h_m, s2h_r, g2h, bb2h);
  // ---- edge path (overwrites hsum scratch region with real edge output) ----
  k_edge<<<NEDGE / 16, 256, 0, stream>>>(x, ea, ei, lng, lnb, W1h, b1e, W2h, b2e, out_e);
  k_colstats<<<(NEDGE + RPB - 1) / RPB, 256, 0, stream>>>(out_e, NEDGE, RPB, s1e_sum, s1e_sq);
  k_bnfinal<<<1, 256, 0, stream>>>(s1e_sum, s1e_sq, 1.0f / NEDGE, s1e_m, s1e_r);
  k_bnapply4<<<4096, 256, 0, stream>>>((const float4*)out_e, (float4*)out_e,
                                       (size_t)NEDGE * DIM / 4, s1e_m, s1e_r, g1e, bb1e);
}
